// SchNetLayer_62835371540676
// MI455X (gfx1250) — matrix operand order repaired and checked by reading, not compiled, not run on hardware
//
#include <hip/hip_runtime.h>

typedef __attribute__((ext_vector_type(16))) _Float16 v16h;
typedef __attribute__((ext_vector_type(8)))  float    v8f;

constexpr int       Dc    = 128;     // feature dim D
constexpr int       Hc    = 256;     // hidden dim H
constexpr int       NODES = 16384;   // B*N
constexpr long long Ec    = 524288;  // edges

// Fast silu: x * rcp(1 + exp(-x)).  v_mul+v_exp+v_add+v_rcp+v_mul, no IEEE div.
__device__ __forceinline__ float silu_f(float x) {
  float e = __expf(-x);
  return x * __builtin_amdgcn_rcpf(1.0f + e);
}

// 16-bit A-matrix 16x32 layout: per-lane K index for packed VGPR pair v (0..7),
// lane-group hi (0 = lanes 0-15, 1 = lanes 16-31); contiguous k = 16*hi + 2*v.
__device__ __forceinline__ int a_kloc(int v, int hi) {
  return 2 * v + hi * 16;
}

// Async bulk copy of a 64KB tile (global f16 image -> LDS) using the CDNA5
// async-to-LDS path (ASYNCcnt).  256 threads x 16 x b128: each instruction
// moves 16B per lane; INST_OFFSET is added to both the LDS and global address.
// Caller must __syncthreads() afterwards (s_wait_asynccnt only covers this
// wave's transfers).
__device__ __forceinline__ void async_stage_64k(void* lds_dst, const void* gsrc,
                                                int tid) {
  unsigned     lbase = (unsigned)(size_t)lds_dst + (unsigned)tid * 16u;
  const char*  gbase = (const char*)gsrc + (size_t)tid * 16u;
  #pragma unroll
  for (int c = 0; c < 16; ++c) {
    asm volatile("global_load_async_to_lds_b128 %0, %1, off offset:%2"
                 :: "v"(lbase), "v"(gbase), "i"(c * 4096) : "memory");
  }
  asm volatile("s_wait_asynccnt 0" ::: "memory");
}

__global__ void k_zero(float* __restrict__ p, int n) {
  int i = blockIdx.x * blockDim.x + threadIdx.x;
  int stride = gridDim.x * blockDim.x;
  for (; i < n; i += stride) p[i] = 0.0f;
}

// One-time weight prep: f32 -> f16, transposed to [n][k] row-major images so
// GEMM blocks can bulk-copy them into LDS with async b128s (no per-block cvt).
__global__ __launch_bounds__(256) void k_prep(
    const float* __restrict__ W2, const float* __restrict__ W3,
    const float* __restrict__ W4,
    _Float16* __restrict__ w2t, _Float16* __restrict__ w3t,
    _Float16* __restrict__ w4t)
{
  int i = blockIdx.x * blockDim.x + threadIdx.x;   // 0 .. 65535
  {
    int n = i >> 8, k = i & 255;
    w3t[i] = (_Float16)W3[k * Hc + n];             // W3: [2D=256, H=256]
  }
  if (i < Dc * Hc) {
    int d = i >> 8, k = i & 255;
    w2t[i] = (_Float16)W2[k * Dc + d];             // W2: [H=256, D=128]
    w4t[i] = (_Float16)W4[k * Dc + d];             // W4: [H=256, D=128]
  }
}

// ---------------------------------------------------------------------------
// Edge kernel: dist -> silu(dist*W1+b1) @ W2 (WMMA f16) -> *h_col -> atomic
// scatter into sums/counts. 8 waves/block, each wave owns 16 edges per iter.
// ---------------------------------------------------------------------------
__global__ __launch_bounds__(256) void k_edges(
    const float* __restrict__ x, const float* __restrict__ h,
    const long long* __restrict__ eidx,
    const float* __restrict__ W1, const float* __restrict__ b1,
    const _Float16* __restrict__ w2t, const float* __restrict__ b2,
    float* __restrict__ sums, float* __restrict__ counts)
{
  __shared__ __align__(32) _Float16 sW2t[Dc][Hc];   // 64KB: W2^T [d][k]

  const int tid = threadIdx.x;
  async_stage_64k(&sW2t[0][0], w2t, tid);
  __syncthreads();

  const int wave = tid >> 5;
  const int lane = tid & 31;
  const int m    = lane & 15;
  const int hi   = (lane >> 4) & 1;

  float b2r[8];
  #pragma unroll
  for (int j = 0; j < 8; ++j) b2r[j] = b2[j * 16 + m];

  constexpr int ITER = 4;
  const long long base = (long long)blockIdx.x * (ITER * 128);

  // One 16-edge tile in flight per wave; W1/b1 stay LICM-hoisted in VGPRs.
  #pragma unroll 1
  for (int it = 0; it < ITER; ++it) {
    const long long e   = base + it * 128 + wave * 16 + m;
    const long long row = eidx[e];
    const long long col = eidx[Ec + e];

    const float dx = x[row * 3 + 0] - x[col * 3 + 0];
    const float dy = x[row * 3 + 1] - x[col * 3 + 1];
    const float dz = x[row * 3 + 2] - x[col * 3 + 2];
    const float dist = sqrtf(dx * dx + dy * dy + dz * dz);

    v8f acc[8] = {};

    #pragma unroll
    for (int s = 0; s < 8; ++s) {
      v16h afrag;
      #pragma unroll
      for (int v = 0; v < 8; ++v) {
        int k = s * 32 + a_kloc(v, hi);
        float2 w1 = *(const float2*)&W1[k];
        float2 bb = *(const float2*)&b1[k];
        afrag[2 * v]     = (_Float16)silu_f(dist * w1.x + bb.x);
        afrag[2 * v + 1] = (_Float16)silu_f(dist * w1.y + bb.y);
      }
      #pragma unroll
      for (int j = 0; j < 8; ++j) {
        int d = j * 16 + m;
        const v16h bfrag = *(const v16h*)&sW2t[d][s * 32 + hi * 16];
        acc[j] = __builtin_amdgcn_wmma_f32_16x16x32_f16(
            false, afrag, false, bfrag, (short)0, acc[j], false, false);
      }
    }

    // Epilogue: C layout row M = r + 8*hi, col N = lane&15.
    #pragma unroll
    for (int r = 0; r < 8; ++r) {
      const long long rw = __shfl(row, r + hi * 8, 16);
      const long long cl = __shfl(col, r + hi * 8, 16);
      #pragma unroll
      for (int j = 0; j < 8; ++j) {
        int d = j * 16 + m;
        float msg = (acc[j][r] + b2r[j]) * h[cl * Dc + d];
        atomicAdd(&sums[rw * Dc + d], msg);
      }
    }
    if (lane < 16) atomicAdd(&counts[row], 1.0f);
  }
}

// ---------------------------------------------------------------------------
// Node update part 1: u = concat(h, agg) @ W3 + b3, LayerNorm, SiLU -> f16 ws.
// W3^T async-staged to LDS in two 64KB halves. 16 nodes per wave.
// ---------------------------------------------------------------------------
__global__ __launch_bounds__(256) void k_update1(
    const float* __restrict__ h, const float* __restrict__ sums,
    const float* __restrict__ counts,
    const _Float16* __restrict__ w3t, const float* __restrict__ b3,
    const float* __restrict__ gam, const float* __restrict__ bet,
    _Float16* __restrict__ un)
{
  __shared__ __align__(32) _Float16 sW3t[128][Hc];  // 64KB: half of W3^T [n][k]

  const int tid  = threadIdx.x;
  const int wave = tid >> 5;
  const int lane = tid & 31;
  const int m    = lane & 15;
  const int hi   = (lane >> 4) & 1;

  const int node0 = blockIdx.x * 128 + wave * 16;
  const long long node = node0 + m;
  const float rc = 1.0f / fmaxf(counts[node], 1.0f);

  v8f acc[16] = {};

  #pragma unroll
  for (int p = 0; p < 2; ++p) {
    __syncthreads();   // all waves done reading previous phase's tile
    async_stage_64k(&sW3t[0][0], w3t + (size_t)p * 128 * Hc, tid);
    __syncthreads();

    #pragma unroll
    for (int s = 0; s < 8; ++s) {
      v16h afrag;
      #pragma unroll
      for (int v = 0; v < 8; ++v) {
        int k = s * 32 + a_kloc(v, hi);
        float a0, a1;
        if (k < Dc) {
          float2 hv = *(const float2*)&h[node * Dc + k];
          a0 = hv.x; a1 = hv.y;
        } else {
          float2 sv = *(const float2*)&sums[node * Dc + (k - Dc)];
          a0 = sv.x * rc; a1 = sv.y * rc;
        }
        afrag[2 * v]     = (_Float16)a0;
        afrag[2 * v + 1] = (_Float16)a1;
      }
      #pragma unroll
      for (int j = 0; j < 8; ++j) {
        int n = j * 16 + m;
        const v16h bfrag = *(const v16h*)&sW3t[n][s * 32 + hi * 16];
        acc[p * 8 + j] = __builtin_amdgcn_wmma_f32_16x16x32_f16(
            false, afrag, false, bfrag, (short)0, acc[p * 8 + j], false, false);
      }
    }
  }

  // acc[jj] covers global column n = (jj/8)*128 + (jj%8)*16 + m.
  float b3r[16];
  #pragma unroll
  for (int j = 0; j < 16; ++j) {
    int n = (j >> 3) * 128 + (j & 7) * 16 + m;
    b3r[j] = b3[n];
  }

  #pragma unroll
  for (int r = 0; r < 8; ++r) {
    float sm = 0.0f, sq = 0.0f;
    #pragma unroll
    for (int j = 0; j < 16; ++j) {
      float u = acc[j][r] + b3r[j];
      acc[j][r] = u;
      sm += u; sq += u * u;
    }
    // Reduce across the 16 lanes of this half-wave (rows live per half-wave).
    #pragma unroll
    for (int o = 1; o < 16; o <<= 1) {
      sm += __shfl_xor(sm, o, 16);
      sq += __shfl_xor(sq, o, 16);
    }
    const float mu  = sm * (1.0f / Hc);
    const float var = sq * (1.0f / Hc) - mu * mu;
    const float rs  = rsqrtf(var + 1e-5f);
    const long long nm = node0 + r + hi * 8;
    #pragma unroll
    for (int j = 0; j < 16; ++j) {
      int n = (j >> 3) * 128 + (j & 7) * 16 + m;
      float unv = (acc[j][r] - mu) * rs * gam[n] + bet[n];
      un[nm * Hc + n] = (_Float16)silu_f(unv);
    }
  }
}

// ---------------------------------------------------------------------------
// Node update part 2: out = silu(un) @ W4 + b4  (un already silu'd, f16).
// ---------------------------------------------------------------------------
__global__ __launch_bounds__(256) void k_update2(
    const _Float16* __restrict__ un,
    const _Float16* __restrict__ w4t, const float* __restrict__ b4,
    float* __restrict__ out)
{
  __shared__ __align__(32) _Float16 sW4t[Dc][Hc];   // 64KB: W4^T [d][k]

  const int tid = threadIdx.x;
  async_stage_64k(&sW4t[0][0], w4t, tid);
  __syncthreads();

  const int wave = tid >> 5, lane = tid & 31, m = lane & 15, hi = (lane >> 4) & 1;
  const int node0 = blockIdx.x * 128 + wave * 16;
  const long long node = node0 + m;

  float b4r[8];
  #pragma unroll
  for (int j = 0; j < 8; ++j) b4r[j] = b4[j * 16 + m];

  v8f acc[8] = {};
  #pragma unroll
  for (int s = 0; s < 8; ++s) {
    v16h afrag;
    #pragma unroll
    for (int v = 0; v < 8; ++v) {
      int k = s * 32 + a_kloc(v, hi);
      afrag[2 * v]     = un[node * Hc + k];
      afrag[2 * v + 1] = un[node * Hc + k + 1];
    }
    #pragma unroll
    for (int j = 0; j < 8; ++j) {
      int d = j * 16 + m;
      const v16h bfrag = *(const v16h*)&sW4t[d][s * 32 + hi * 16];
      acc[j] = __builtin_amdgcn_wmma_f32_16x16x32_f16(
          false, afrag, false, bfrag, (short)0, acc[j], false, false);
    }
  }

  #pragma unroll
  for (int r = 0; r < 8; ++r) {
    const long long nm = node0 + r + hi * 8;
    #pragma unroll
    for (int j = 0; j < 8; ++j) {
      out[nm * Dc + j * 16 + m] = acc[j][r] + b4r[j];
    }
  }
}

extern "C" void kernel_launch(void* const* d_in, const int* in_sizes, int n_in,
                              void* d_out, int out_size, void* d_ws, size_t ws_size,
                              hipStream_t stream) {
  const float*     x    = (const float*)d_in[0];
  const float*     h    = (const float*)d_in[1];
  const long long* eidx = (const long long*)d_in[2];
  // d_in[3] = batch_size (unused, compile-time constant)
  const float* W1  = (const float*)d_in[4];
  const float* b1  = (const float*)d_in[5];
  const float* W2  = (const float*)d_in[6];
  const float* b2  = (const float*)d_in[7];
  const float* W3  = (const float*)d_in[8];
  const float* b3  = (const float*)d_in[9];
  const float* gam = (const float*)d_in[10];
  const float* bet = (const float*)d_in[11];
  const float* W4  = (const float*)d_in[12];
  const float* b4  = (const float*)d_in[13];

  // Workspace: sums [16384*128 f32] | counts [16384 f32] | un [16384*256 f16]
  //          | w2t [128*256 f16] | w3t [256*256 f16] | w4t [128*256 f16]
  float*     sums   = (float*)d_ws;
  float*     counts = sums + (size_t)NODES * Dc;
  _Float16*  un     = (_Float16*)(counts + NODES);
  _Float16*  w2t    = un + (size_t)NODES * Hc;
  _Float16*  w3t    = w2t + (size_t)Dc * Hc;
  _Float16*  w4t    = w3t + (size_t)Hc * Hc;
  float*     out    = (float*)d_out;

  const int zero_n = NODES * Dc + NODES;
  hipLaunchKernelGGL(k_zero, dim3(1024), dim3(256), 0, stream, sums, zero_n);
  hipLaunchKernelGGL(k_prep, dim3((Hc * Hc) / 256), dim3(256), 0, stream,
                     W2, W3, W4, w2t, w3t, w4t);
  hipLaunchKernelGGL(k_edges, dim3((int)(Ec / 512)), dim3(256), 0, stream,
                     x, h, eidx, W1, b1, w2t, b2, sums, counts);
  hipLaunchKernelGGL(k_update1, dim3(NODES / 128), dim3(256), 0, stream,
                     h, sums, counts, w3t, b3, gam, bet, un);
  hipLaunchKernelGGL(k_update2, dim3(NODES / 128), dim3(256), 0, stream,
                     un, w4t, b4, out);
}
